// LSTM_proj_4372276707579
// MI455X (gfx1250) — matrix operand order repaired and checked by reading, not compiled, not run on hardware
//
#include <hip/hip_runtime.h>

// Problem constants (from reference)
#define IN_CH 9
#define HID   256
#define NF    1024          // sequence length
#define BTOT  2048          // batch
#define G4    1024          // 4*HID gate width

typedef __attribute__((ext_vector_type(16))) _Float16 v16h;
typedef __attribute__((ext_vector_type(8)))  float    v8f;

// ---------------------------------------------------------------------------
// Kernel 1: A[b,g] = sum_{d<9} x[b,d] * W_ih0[g, 1+d]  + (b_ih0[g]+b_hh0[g])
// One wave per 16x16 output tile, V_WMMA_F32_16X16X32_F16 (K=9 padded to 32).
// A-matrix lane layout (ISA 7.12.2, 16-bit A 16x32):
//   lane L: M = L%16;  element e<8  -> K = e + 8*(L>=16)
//                      element e>=8 -> K = e + 8 + 8*(L>=16)
// B-matrix (32x16, 16-bit): lane L: N = L%16; element e -> same K as A element e
// C/D (f32 16x16): lane L: N = L%16; VGPR r -> M = r + 8*(L>=16)
// ---------------------------------------------------------------------------
__global__ __launch_bounds__(32) void precomp_A_wmma(
    const float* __restrict__ x,       // [BTOT, IN_CH]
    const float* __restrict__ Wih0,    // [G4, IN_CH+1]
    const float* __restrict__ bih0,    // layer-0 slice [G4]
    const float* __restrict__ bhh0,    // layer-0 slice [G4]
    float* __restrict__ A)             // [BTOT, G4]
{
    const int lane = threadIdx.x & 31;
    const int b0 = blockIdx.x * 16;     // batch tile
    const int g0 = blockIdx.y * 16;     // gate tile
    const int hi = lane >> 4;
    const int mn = lane & 15;

    v16h av, bv;
#pragma unroll
    for (int e = 0; e < 16; ++e) {
        int ka = e + 8 * hi + ((e >= 8) ? 8 : 0);          // A element K index
        av[e] = (ka < IN_CH) ? (_Float16)x[(size_t)(b0 + mn) * IN_CH + ka]
                             : (_Float16)0.0f;
        int kb = e + 8 * hi + ((e >= 8) ? 8 : 0);          // B element K index
        bv[e] = (kb < IN_CH) ? (_Float16)Wih0[(size_t)(g0 + mn) * (IN_CH + 1) + 1 + kb]
                             : (_Float16)0.0f;
    }

    v8f c = {};
    // (neg_a, A, neg_b, B, c_mod, C, reuse_a, reuse_b)
    c = __builtin_amdgcn_wmma_f32_16x16x32_f16(false, av, false, bv,
                                               (short)0, c, false, false);

    const float bias = bih0[g0 + mn] + bhh0[g0 + mn];      // depends on column only
#pragma unroll
    for (int r = 0; r < 8; ++r) {
        int m = r + 8 * hi;
        A[(size_t)(b0 + m) * G4 + (g0 + mn)] = c[r] + bias;
    }
}

// ---------------------------------------------------------------------------
// Kernel 2: one proj-LSTM layer, one wave32 per batch element.
// Lane l owns hidden channels j = l + 32*r, r = 0..7 (gate blocks i/f/g/o are
// register groups k=0..7 / 8..15 / 16..23 / 24..31 with j_k = l + 32*k).
// Per step: gates = base + s*wih + h*whh ; LSTM cell ; h = <sigma(o)*tanh(c), whr>
// reduced across the wave with shfl_xor (no LDS, no barriers).
//   mode 0 (layer 0):  base = A row (per batch),   s = t/(NF-1)
//   mode 1 (layers 1,2): base = bias (shared),     s = u[b,t] (prev layer out)
// ---------------------------------------------------------------------------
__device__ __forceinline__ float fast_sig(float x) {
    return __builtin_amdgcn_rcpf(1.0f + __expf(-x));
}
__device__ __forceinline__ float fast_tanh(float x) {
    float e = __expf(-2.0f * x);
    return (1.0f - e) * __builtin_amdgcn_rcpf(1.0f + e);
}

__global__ __launch_bounds__(256) void lstm_scan(
    int mode,
    const float* __restrict__ Abase,   // [BTOT, G4]   (mode 0)
    const float* __restrict__ uin,     // [BTOT, NF]   (mode 1)
    const float* __restrict__ wih, int wih_stride,  // column vector, strided
    const float* __restrict__ whh,     // [G4] layer slice
    const float* __restrict__ bias_i,  // [G4] layer slice (mode 1)
    const float* __restrict__ bias_h,  // [G4] layer slice (mode 1)
    const float* __restrict__ whr,     // [HID] layer slice
    float* __restrict__ hout)          // [BTOT, NF]
{
    const int lane = threadIdx.x & 31;
    const int wid  = threadIdx.x >> 5;
    const int b    = blockIdx.x * (blockDim.x >> 5) + wid;

    // Pin everything in VGPRs: 32+32+32+8+8 floats per lane.
    float base[32], w_in[32], w_h[32], w_r[8], c[8];
#pragma unroll
    for (int k = 0; k < 32; ++k) {
        const int j = lane + 32 * k;
        w_in[k] = wih[(size_t)j * wih_stride];
        w_h[k]  = whh[j];
        if (mode == 0) base[k] = Abase[(size_t)b * G4 + j];
        else           base[k] = bias_i[j] + bias_h[j];
    }
#pragma unroll
    for (int r = 0; r < 8; ++r) { w_r[r] = whr[lane + 32 * r]; c[r] = 0.0f; }

    const float inv = 1.0f / (float)(NF - 1);
    float h = 0.0f;
    float s_cur = (mode == 0) ? 0.0f : uin[(size_t)b * NF];

    for (int t = 0; t < NF; ++t) {
        // Prefetch next step's input scalar before the dependent chain.
        float s_next = 0.0f;
        if (mode != 0 && (t + 1) < NF) s_next = uin[(size_t)b * NF + t + 1];
        const float s = (mode == 0) ? (float)t * inv : s_cur;

        float part = 0.0f;
#pragma unroll
        for (int r = 0; r < 8; ++r) {
            float gi = fmaf(h, w_h[r],      fmaf(s, w_in[r],      base[r]));
            float gf = fmaf(h, w_h[8 + r],  fmaf(s, w_in[8 + r],  base[8 + r]));
            float gg = fmaf(h, w_h[16 + r], fmaf(s, w_in[16 + r], base[16 + r]));
            float go = fmaf(h, w_h[24 + r], fmaf(s, w_in[24 + r], base[24 + r]));
            float ii = fast_sig(gi);
            float ff = fast_sig(gf);
            float tg = fast_tanh(gg);
            float oo = fast_sig(go);
            c[r] = fmaf(ff, c[r], ii * tg);
            part = fmaf(oo * fast_tanh(c[r]), w_r[r], part);
        }
        // wave32 reduction -> projected scalar h
#pragma unroll
        for (int off = 16; off > 0; off >>= 1)
            part += __shfl_xor(part, off, 32);
        h = part;

        if (lane == 0) hout[(size_t)b * NF + t] = h;
        s_cur = s_next;
    }
}

// ---------------------------------------------------------------------------
// Host-side launch. Layers are serialized on `stream`:
//   precomp A (WMMA) -> layer0 (out as h0 buf) -> layer1 (ws) -> layer2 (out)
// ---------------------------------------------------------------------------
extern "C" void kernel_launch(void* const* d_in, const int* in_sizes, int n_in,
                              void* d_out, int out_size, void* d_ws, size_t ws_size,
                              hipStream_t stream) {
    (void)in_sizes; (void)n_in; (void)out_size; (void)ws_size;

    const float* x    = (const float*)d_in[0];  // [2048, 9]
    const float* Wih0 = (const float*)d_in[1];  // [1024, 10]
    const float* Wih1 = (const float*)d_in[2];  // [1024, 1]
    const float* Wih2 = (const float*)d_in[3];  // [1024, 1]
    const float* Whh  = (const float*)d_in[4];  // [3, 1024, 1]
    const float* bih  = (const float*)d_in[5];  // [3, 1024]
    const float* bhh  = (const float*)d_in[6];  // [3, 1024]
    const float* Whr  = (const float*)d_in[7];  // [3, 1, 256]
    float* out = (float*)d_out;                 // [2048, 1, 1024] == [B, NF]

    float* A  = (float*)d_ws;                                   // 8 MB
    float* h1 = (float*)((char*)d_ws + (size_t)BTOT * G4 * sizeof(float)); // 8 MB

    // 1) A = x @ W_ih0[:,1:].T + (b_ih0 + b_hh0), via WMMA
    dim3 gA(BTOT / 16, G4 / 16);
    precomp_A_wmma<<<gA, 32, 0, stream>>>(x, Wih0, bih, bhh, A);

    // 2) Sequential scans, one wave per batch element (8 waves / 256-thread block)
    dim3 gS(BTOT / 8);
    // layer 0: wih = W_ih0[:,0] (stride 10); bias folded into A; h0 -> out
    lstm_scan<<<gS, 256, 0, stream>>>(0, A, out, Wih0, IN_CH + 1,
                                      Whh + 0 * G4, bih + 0 * G4, bhh + 0 * G4,
                                      Whr + 0 * HID, out);
    // layer 1: reads out, writes h1
    lstm_scan<<<gS, 256, 0, stream>>>(1, A, out, Wih1, 1,
                                      Whh + 1 * G4, bih + 1 * G4, bhh + 1 * G4,
                                      Whr + 1 * HID, h1);
    // layer 2: reads h1, writes final out
    lstm_scan<<<gS, 256, 0, stream>>>(1, A, h1, Wih2, 1,
                                      Whh + 2 * G4, bih + 2 * G4, bhh + 2 * G4,
                                      Whr + 2 * HID, out);
}
